// FlaxLMUFFT_43671227466478
// MI455X (gfx1250) — compile-verified, matrix operand order checked by reading, not run on hardware
//
#include <hip/hip_runtime.h>
#include <hip/hip_bf16.h>

typedef _Float16 half_t;
typedef __attribute__((ext_vector_type(16))) _Float16 v16h;
typedef __attribute__((ext_vector_type(8)))  float    v8f;

#define BATCH 64
#define SEQ   512
#define DIN   512
#define MEM   256
#define HID   512
#define KTOT  768                 // MEM + DIN
#define ROWS  (BATCH * SEQ)       // 32768

// ---------------------------------------------------------------------------
// u[b,t] = relu( x[b,t,:] . Wu_w + Wu_b )   — one wave32 per row
// ---------------------------------------------------------------------------
__global__ __launch_bounds__(256) void k_u(const float* __restrict__ x,
                                           const float* __restrict__ Wu_w,
                                           const float* __restrict__ Wu_b,
                                           float* __restrict__ u) {
    const int lane = threadIdx.x & 31;
    const int wave = threadIdx.x >> 5;
    const int row  = blockIdx.x * 8 + wave;       // 4096 blocks * 8 waves = 32768
    const float* xr = x + (size_t)row * DIN;
    float p = 0.f;
    #pragma unroll
    for (int i = 0; i < DIN / 32; ++i)
        p += xr[lane + 32 * i] * Wu_w[lane + 32 * i];
    #pragma unroll
    for (int off = 16; off > 0; off >>= 1)
        p += __shfl_xor(p, off, 32);
    if (lane == 0)
        u[row] = fmaxf(p + Wu_b[0], 0.f);
}

// ---------------------------------------------------------------------------
// 256x256 f32 matmul: C = A @ B  (used for repeated squaring A^2..A^32)
// ---------------------------------------------------------------------------
__global__ __launch_bounds__(256) void k_mm256(const float* __restrict__ A,
                                               const float* __restrict__ B,
                                               float* __restrict__ C) {
    const int i = blockIdx.x;          // row
    const int j = threadIdx.x;         // col
    float acc = 0.f;
    for (int k = 0; k < 256; ++k)
        acc += A[i * 256 + k] * B[k * 256 + j];
    C[i * 256 + j] = acc;
}

// 256x256 transpose: out[c][r] = in[r][c]
__global__ __launch_bounds__(256) void k_tr(const float* __restrict__ in,
                                            float* __restrict__ out) {
    int idx = blockIdx.x * 256 + threadIdx.x;     // 256 blocks
    int r = idx >> 8, c = idx & 255;
    out[c * 256 + r] = in[r * 256 + c];
}

// ---------------------------------------------------------------------------
// seeds[g] = A^(32g) @ B, g=0..15 — one block, sequential with P32^T coalesced
// ---------------------------------------------------------------------------
__global__ __launch_bounds__(256) void k_seeds(const float* __restrict__ P32T,
                                               const float* __restrict__ Bvec,
                                               float* __restrict__ seeds) {
    __shared__ float st[256];
    const int k = threadIdx.x;
    st[k] = Bvec[k];
    seeds[k] = st[k];
    __syncthreads();
    for (int g = 1; g < 16; ++g) {
        float v = 0.f;
        for (int j = 0; j < 256; ++j)
            v += P32T[j * 256 + k] * st[j];        // coalesced across k
        __syncthreads();
        st[k] = v;
        seeds[g * 256 + k] = v;
        __syncthreads();
    }
}

// ---------------------------------------------------------------------------
// Hh[k][s] = (A^s B)[k] as f16, 16 blocks each produce 32 columns from a seed
// ---------------------------------------------------------------------------
__global__ __launch_bounds__(256) void k_hh(const float* __restrict__ At,   // A^T
                                            const float* __restrict__ seeds,
                                            half_t* __restrict__ Hh) {
    __shared__ float st[256];
    const int g = blockIdx.x;
    const int k = threadIdx.x;
    st[k] = seeds[g * 256 + k];
    __syncthreads();
    for (int t = 0; t < 32; ++t) {
        Hh[k * SEQ + (g * 32 + t)] = (half_t)st[k];
        float v = 0.f;
        for (int j = 0; j < 256; ++j)
            v += At[j * 256 + k] * st[j];          // coalesced across k
        __syncthreads();
        st[k] = v;
        __syncthreads();
    }
}

// ---------------------------------------------------------------------------
// x (f32) -> Af columns [256..768) as f16 (row-major, pitch 768)
// ---------------------------------------------------------------------------
__global__ __launch_bounds__(256) void k_cvt_x(const float* __restrict__ x,
                                               half_t* __restrict__ Af) {
    const size_t stride = (size_t)gridDim.x * blockDim.x;
    for (size_t idx = blockIdx.x * (size_t)blockDim.x + threadIdx.x;
         idx < (size_t)ROWS * DIN; idx += stride) {
        size_t row = idx >> 9;       // /512
        size_t d   = idx & 511;
        Af[row * KTOT + MEM + d] = (half_t)x[idx];
    }
}

// Wh_w (768x512 f32, K-major) -> WhT (512x768 f16, N-major: WhT[n][k])
__global__ __launch_bounds__(256) void k_cvt_wht(const float* __restrict__ Wh,
                                                 half_t* __restrict__ WhT) {
    const size_t stride = (size_t)gridDim.x * blockDim.x;
    for (size_t idx = blockIdx.x * (size_t)blockDim.x + threadIdx.x;
         idx < (size_t)KTOT * HID; idx += stride) {
        size_t n = idx / KTOT;
        size_t k = idx - n * KTOT;
        WhT[idx] = (half_t)Wh[k * HID + n];
    }
}

// ---------------------------------------------------------------------------
// 16-bit WMMA operand fragment from row/N-major f16, K contiguous.
// lanes 0-15 hold K {kk..kk+7, kk+16..kk+23}; lanes 16-31 offset +8.
// ---------------------------------------------------------------------------
__device__ __forceinline__ v16h load_frag(const half_t* __restrict__ base,
                                          size_t row, int pitch, int kk, int lhalf) {
    const half_t* p = base + row * (size_t)pitch + kk + lhalf * 8;
    v16h v;
    #pragma unroll
    for (int i = 0; i < 8; ++i) { v[i] = p[i]; v[i + 8] = p[16 + i]; }
    return v;
}

// ---------------------------------------------------------------------------
// Conv-as-GEMM (WMMA): per batch b, m_b(512x256) = Toeplitz(u_b) @ Hh^T
// Toeplitz A-fragments gathered branch-free from a reversed, zero-padded
// copy of u_b in LDS: uR[x] = u[511-x] (x<=511), uR[512..543]=0, so
// T[t,d]=u[t-d] == uR[(511-t+dbase)+i] — two contiguous 8-half runs per lane.
// Lower-triangular => only kk <= Morig tiles contribute.
// grid (4,4,64), block 256 (8 waves, 4x2, 32x32 tile each)
// ---------------------------------------------------------------------------
__global__ __launch_bounds__(256) void k_conv(const float* __restrict__ u,
                                              const half_t* __restrict__ Hh,
                                              half_t* __restrict__ Af) {
    __shared__ half_t uR[SEQ + 32];
    const int b = blockIdx.z;
    for (int i = threadIdx.x; i < SEQ + 32; i += 256)
        uR[i] = (i <= SEQ - 1) ? (half_t)u[b * SEQ + (SEQ - 1 - i)] : (half_t)0;
    __syncthreads();

    const int wave  = threadIdx.x >> 5;
    const int lane  = threadIdx.x & 31;
    const int wm    = wave & 3, wn = wave >> 2;
    const int Morig = blockIdx.x * 128 + wm * 32;      // t-tile
    const int Norig = blockIdx.y * 64  + wn * 32;      // mem-tile
    const int lhalf = lane >> 4;
    const int lrow  = lane & 15;

    v8f acc[2][2] = {};
    for (int kk = 0; kk <= Morig; kk += 32) {          // triangular K range
        v16h a[2], bf[2];
        #pragma unroll
        for (int f = 0; f < 2; ++f) {
            const int t = Morig + f * 16 + lrow;
            const int q = (SEQ - 1) - t + kk + lhalf * 8;  // >= 0 within range
            v16h av;
            #pragma unroll
            for (int i = 0; i < 8; ++i) {
                av[i]     = uR[q + i];
                av[i + 8] = uR[q + 16 + i];
            }
            a[f]  = av;
            bf[f] = load_frag(Hh, (size_t)(Norig + f * 16 + lrow), SEQ, kk, lhalf);
        }
        #pragma unroll
        for (int i = 0; i < 2; ++i)
            #pragma unroll
            for (int j = 0; j < 2; ++j)
                acc[i][j] = __builtin_amdgcn_wmma_f32_16x16x32_f16(
                    false, a[i], false, bf[j], (short)0, acc[i][j], false, false);
    }
    #pragma unroll
    for (int i = 0; i < 2; ++i)
        #pragma unroll
        for (int j = 0; j < 2; ++j)
            #pragma unroll
            for (int v = 0; v < 8; ++v) {
                const int t = Morig + i * 16 + v + 8 * lhalf;
                const int c = Norig + j * 16 + lrow;
                Af[(size_t)(b * SEQ + t) * KTOT + c] = (half_t)acc[i][j][v];
            }
}

// ---------------------------------------------------------------------------
// Main GEMM (WMMA): h = relu(Af(32768x768) @ Wh + bias); also emits h_n rows.
// grid (256,8), block 256 (8 waves, 4x2, 32x32 tile each)
// ---------------------------------------------------------------------------
__global__ __launch_bounds__(256) void k_gemm(const half_t* __restrict__ Af,
                                              const half_t* __restrict__ WhT,
                                              const float* __restrict__ bias,
                                              float* __restrict__ h,
                                              float* __restrict__ hn) {
    const int wave  = threadIdx.x >> 5;
    const int lane  = threadIdx.x & 31;
    const int wm    = wave & 3, wn = wave >> 2;
    const int Morig = blockIdx.x * 128 + wm * 32;
    const int Norig = blockIdx.y * 64  + wn * 32;
    const int lhalf = lane >> 4;
    const int lrow  = lane & 15;

    v8f acc[2][2] = {};
    for (int kk = 0; kk < KTOT; kk += 32) {
        v16h a[2], bf[2];
        #pragma unroll
        for (int f = 0; f < 2; ++f) {
            a[f]  = load_frag(Af,  (size_t)(Morig + f * 16 + lrow), KTOT, kk, lhalf);
            bf[f] = load_frag(WhT, (size_t)(Norig + f * 16 + lrow), KTOT, kk, lhalf);
        }
        #pragma unroll
        for (int i = 0; i < 2; ++i)
            #pragma unroll
            for (int j = 0; j < 2; ++j)
                acc[i][j] = __builtin_amdgcn_wmma_f32_16x16x32_f16(
                    false, a[i], false, bf[j], (short)0, acc[i][j], false, false);
    }
    #pragma unroll
    for (int i = 0; i < 2; ++i)
        #pragma unroll
        for (int j = 0; j < 2; ++j) {
            const int c  = Norig + j * 16 + lrow;
            const float bv = bias[c];
            #pragma unroll
            for (int v = 0; v < 8; ++v) {
                const int r = Morig + i * 16 + v + 8 * lhalf;
                const float val = fmaxf(acc[i][j][v] + bv, 0.f);
                h[(size_t)r * HID + c] = val;
                if ((r & (SEQ - 1)) == SEQ - 1)          // t == SEQ-1
                    hn[(size_t)(r >> 9) * HID + c] = val;
            }
        }
}

// ---------------------------------------------------------------------------
extern "C" void kernel_launch(void* const* d_in, const int* in_sizes, int n_in,
                              void* d_out, int out_size, void* d_ws, size_t ws_size,
                              hipStream_t stream) {
    const float* x    = (const float*)d_in[0];
    const float* Wu_w = (const float*)d_in[1];
    const float* Wu_b = (const float*)d_in[2];
    const float* Wh_w = (const float*)d_in[3];
    const float* Wh_b = (const float*)d_in[4];
    const float* A    = (const float*)d_in[5];
    const float* Bv   = (const float*)d_in[6];

    // workspace carve-up (~52.3 MB total)
    char* ws = (char*)d_ws;
    float*  u     = (float*)(ws);                         size_t off = (size_t)ROWS * 4;
    float*  At    = (float*)(ws + off);                   off += 65536 * 4;
    float*  Pa    = (float*)(ws + off);                   off += 65536 * 4;
    float*  Pb    = (float*)(ws + off);                   off += 65536 * 4;
    float*  seeds = (float*)(ws + off);                   off += 16 * 256 * 4;
    half_t* Hh    = (half_t*)(ws + off);                  off += (size_t)MEM * SEQ * 2;
    half_t* WhT   = (half_t*)(ws + off);                  off += (size_t)HID * KTOT * 2;
    half_t* Af    = (half_t*)(ws + off);                  // ROWS*KTOT f16 = 48 MB

    float* h  = (float*)d_out;
    float* hn = h + (size_t)ROWS * HID;

    // 1) u = relu(x @ Wu_w + b)
    k_u<<<ROWS / 8, 256, 0, stream>>>(x, Wu_w, Wu_b, u);

    // 2) A^32 by repeated squaring; transposes for coalesced mat-vec chains
    k_mm256<<<256, 256, 0, stream>>>(A,  A,  Pa);   // A^2
    k_mm256<<<256, 256, 0, stream>>>(Pa, Pa, Pb);   // A^4
    k_mm256<<<256, 256, 0, stream>>>(Pb, Pb, Pa);   // A^8
    k_mm256<<<256, 256, 0, stream>>>(Pa, Pa, Pb);   // A^16
    k_mm256<<<256, 256, 0, stream>>>(Pb, Pb, Pa);   // A^32
    k_tr<<<256, 256, 0, stream>>>(Pa, Pb);          // (A^32)^T
    k_tr<<<256, 256, 0, stream>>>(A,  At);          // A^T

    // 3) seeds and the 16-way-parallel impulse response H (f16)
    k_seeds<<<1, 256, 0, stream>>>(Pb, Bv, seeds);
    k_hh<<<16, 256, 0, stream>>>(At, seeds, Hh);

    // 4) pack concat(m, x) A-matrix and Wh^T in f16
    k_cvt_x<<<4096, 256, 0, stream>>>(x, Af);
    k_cvt_wht<<<1536, 256, 0, stream>>>(Wh_w, WhT);

    // 5) WMMA Toeplitz conv -> m columns of Af
    k_conv<<<dim3(SEQ / 128, MEM / 64, BATCH), 256, 0, stream>>>(u, Hh, Af);

    // 6) WMMA main GEMM + bias + relu + h_n extraction
    k_gemm<<<dim3(ROWS / 128, HID / 64), 256, 0, stream>>>(Af, WhT, Wh_b, h, hn);
    (void)in_sizes; (void)n_in; (void)out_size; (void)ws_size;
}
